// SE3PointNeighborConvolution_60593398612079
// MI455X (gfx1250) — matrix operand order repaired and checked by reading, not compiled, not run on hardware
//
#include <hip/hip_runtime.h>
#include <hip/hip_bf16.h>

typedef __attribute__((ext_vector_type(16))) __bf16 v16bf;
typedef __attribute__((ext_vector_type(8)))  __bf16 v8bf;
typedef __attribute__((ext_vector_type(8)))  float  v8f;

#define NNB    4        // n
#define NB     8192     // b
#define CIN    32
#define DOUT   32
#define NA     32       // neighbors
#define NK     36       // 4 radii * 9 SH
#define TILE_P 16
#define KROWS  48       // basisT k-rows allocated (only 0..35 valid)

// Padded strides (all 16B-aligned, chosen so lane-stride dwords % 64 != 0)
#define ASTR   40                   // a-row stride for featT/basisT (20 dw: conflict-free frags)
#define MPSTR  1160                 // Mflat per-point stride (580 dw = 4 mod 64: conflict-free)
#define WDSTR  40                   // Wlds d stride
#define WKSTR  1288                 // Wlds k stride (644 dw = 4 mod 64: 2-way staging stores)

#define REGA_ELEMS (TILE_P*CIN*ASTR + TILE_P*KROWS*ASTR)   // 51200 (>= 36*WKSTR = 46368)

// two contiguous 16B LDS reads -> one 16-element bf16 fragment
__device__ __forceinline__ v16bf ld_frag(const __bf16* p0, const __bf16* p1) {
    union { v16bf v; v8bf h[2]; } u;
    u.h[0] = *(const v8bf*)p0;
    u.h[1] = *(const v8bf*)p1;
    return u.v;
}

__global__ __launch_bounds__(128)
void se3_conv_kernel(const float* __restrict__ input,
                     const float* __restrict__ coords,
                     const float* __restrict__ mask,
                     const float* __restrict__ W,
                     const int*   __restrict__ neighbors,
                     float* __restrict__ out)
{
    // Region A (100KB): featT[p][c][a] + basisT[p][k][a] (phases A/B),
    // reused as Wlds[k][d][c] (90.6KB) in phase C.
    __shared__ __align__(16) __bf16 smem[REGA_ELEMS];
    __bf16* featT  = smem;                          // [(p*CIN+c)*ASTR + a]
    __bf16* basisT = smem + TILE_P * CIN * ASTR;    // [(p*KROWS+k)*ASTR + a]
    __bf16* Wlds   = smem;                          // [k*WKSTR + d*WDSTR + c]
    __shared__ __align__(16) __bf16 Mflat[TILE_P * MPSTR];  // [p*MPSTR + k*32 + c]
    __shared__ float partial[4][TILE_P][DOUT];

    const int tid  = threadIdx.x;
    const int lane = tid & 31;
    const int l15  = lane & 15;
    const int hi   = (lane >> 4) & 1;
    const int wv   = tid >> 5;                 // wave 0..3 (wave32)
    const int g0   = blockIdx.x * TILE_P;
    const int nn   = g0 / NB;                  // 8192 % 16 == 0 -> uniform per tile
    const int bb0  = g0 % NB;

    // ---------------- Phase A: gather features + build (mask-folded) basis ----------------
    #pragma unroll
    for (int it = 0; it < 4; ++it) {
        int pair = tid + 128 * it;             // 512 (p,a) pairs
        int p = pair >> 5;
        int a = pair & 31;
        int bb = bb0 + p;
        int base = nn * NB + bb;
        int nidx = neighbors[base * NA + a];
        float m  = mask[base * NA + a];
        const float* cs = coords + (size_t)base * 3;
        const float* cg = coords + (size_t)(nn * NB + nidx) * 3;
        float dx = cg[0] - cs[0], dy = cg[1] - cs[1], dz = cg[2] - cs[2];
        float r   = sqrtf(dx*dx + dy*dy + dz*dz);
        float inv = 1.0f / (r + 1e-8f);
        float x = dx * inv, y = dy * inv, z = dz * inv;

        float Y[9];
        Y[0] = 0.28209479f;
        Y[1] = 0.48860251f * x;
        Y[2] = 0.48860251f * y;
        Y[3] = 0.48860251f * z;
        Y[4] = 1.09254843f * x * y;
        Y[5] = 1.09254843f * y * z;
        Y[6] = 0.31539157f * (3.0f * z * z - 1.0f);
        Y[7] = 1.09254843f * x * z;
        Y[8] = 0.54627421f * (x * x - y * y);

        const float R[4] = {0.5f, 1.0f, 1.5f, 2.0f};
        #pragma unroll
        for (int ri = 0; ri < 4; ++ri) {
            float t   = (r - R[ri]) * 2.0f;    // (r-R)/sigma, sigma=0.5
            float rbm = m * __expf(-t * t);    // fold relative_mask into basis
            #pragma unroll
            for (int sh = 0; sh < 9; ++sh)
                basisT[(p * KROWS + ri * 9 + sh) * ASTR + a] = (__bf16)(rbm * Y[sh]);
        }
        const float* isrc = input + (size_t)nn * CIN * NB + nidx;
        #pragma unroll
        for (int c = 0; c < CIN; ++c)
            featT[(p * CIN + c) * ASTR + a] = (__bf16)isrc[(size_t)c * NB];
    }
    __syncthreads();

    // ---------------- Phase B: M[c,k] = feat^T x basis (wave owns 4 points) ----------------
    for (int pi = 0; pi < 4; ++pi) {
        int p = wv * 4 + pi;
        v16bf A[2], B[3];
        #pragma unroll
        for (int ct = 0; ct < 2; ++ct) {       // A: row=c, K=a
            const __bf16* b = &featT[(p * CIN + ct * 16 + l15) * ASTR];
            A[ct] = ld_frag(b + 8 * hi, b + 16 + 8 * hi);
        }
        #pragma unroll
        for (int kt = 0; kt < 3; ++kt) {       // B: col=k, K=a
            const __bf16* b = &basisT[(p * KROWS + kt * 16 + l15) * ASTR];
            B[kt] = ld_frag(b + 8 * hi, b + 16 + 8 * hi);
        }
        v8f acc[2][3];
        #pragma unroll
        for (int ct = 0; ct < 2; ++ct)
            #pragma unroll
            for (int kt = 0; kt < 3; ++kt) {
                v8f zero = {};
                acc[ct][kt] = __builtin_amdgcn_wmma_f32_16x16x32_bf16(
                    false, A[ct], false, B[kt], (short)0, zero, false, false);
            }
        // store M k-major: kk = k*32 + c; 8 rows per lane contiguous -> b128 store
        #pragma unroll
        for (int ct = 0; ct < 2; ++ct)
            #pragma unroll
            for (int kt = 0; kt < 3; ++kt) {
                int k = kt * 16 + l15;
                v8bf pk;
                #pragma unroll
                for (int i = 0; i < 8; ++i) pk[i] = (__bf16)acc[ct][kt][i];
                if (k < NK)
                    *(v8bf*)&Mflat[p * MPSTR + k * CIN + ct * 16 + 8 * hi] = pk;
            }
    }
    __syncthreads();

    // ---------------- stage W -> LDS bf16, Wlds[k][d][c] (reuses feat/basis space) ----------------
    #pragma unroll 4
    for (int i = 0; i < (DOUT * CIN * NK) / 128; ++i) {   // 288 iters, coalesced global reads
        int o   = tid + 128 * i;
        int d   = o / (CIN * NK);
        int rem = o - d * (CIN * NK);
        int c   = rem / NK;
        int k   = rem - c * NK;
        Wlds[k * WKSTR + d * WDSTR + c] = (__bf16)W[o];
    }
    __syncthreads();

    // ---------------- Phase C: out[16p x 32d] = M[16 x 1152] x W^T, K split over waves ----------------
    v8f dacc[2] = {};
    for (int t = wv * 9; t < wv * 9 + 9; ++t) {          // 36 k-tiles / 4 waves
        const __bf16* ab = &Mflat[l15 * MPSTR + t * 32 + 8 * hi];
        v16bf Af = ld_frag(ab, ab + 16);                 // A: row=point, K=kk
        #pragma unroll
        for (int nt = 0; nt < 2; ++nt) {                 // B: col=d, K=kk (k=t, c=8hi+j / 16+8hi+j-8)
            const __bf16* bb = &Wlds[t * WKSTR + (nt * 16 + l15) * WDSTR];
            v16bf Bf = ld_frag(bb + 8 * hi, bb + 16 + 8 * hi);
            dacc[nt] = __builtin_amdgcn_wmma_f32_16x16x32_bf16(
                false, Af, false, Bf, (short)0, dacc[nt], false, false);
        }
    }
    #pragma unroll
    for (int nt = 0; nt < 2; ++nt)
        #pragma unroll
        for (int i = 0; i < 8; ++i)
            partial[wv][i + 8 * hi][nt * 16 + l15] = dacc[nt][i];
    __syncthreads();

    // reduce wave partials, write out[n,d,b]
    #pragma unroll
    for (int it = 0; it < 4; ++it) {
        int o = tid + 128 * it;               // 512 outputs
        int p = o >> 5;
        int d = o & 31;
        float s = partial[0][p][d] + partial[1][p][d]
                + partial[2][p][d] + partial[3][p][d];
        out[((size_t)nn * DOUT + d) * NB + (bb0 + p)] = s;
    }
}

extern "C" void kernel_launch(void* const* d_in, const int* in_sizes, int n_in,
                              void* d_out, int out_size, void* d_ws, size_t ws_size,
                              hipStream_t stream) {
    const float* input     = (const float*)d_in[0];
    const float* coords    = (const float*)d_in[1];
    const float* mask      = (const float*)d_in[2];
    const float* W         = (const float*)d_in[3];
    const int*   neighbors = (const int*)d_in[4];
    float* out = (float*)d_out;

    const int n_points = NNB * NB;            // 32768
    const int blocks   = n_points / TILE_P;   // 2048
    se3_conv_kernel<<<blocks, 128, 0, stream>>>(input, coords, mask, W, neighbors, out);
    (void)in_sizes; (void)n_in; (void)out_size; (void)d_ws; (void)ws_size;
}
